// Attention_51874615001302
// MI455X (gfx1250) — matrix operand order repaired and checked by reading, not compiled, not run on hardware
//
#include <hip/hip_runtime.h>

// ---------------------------------------------------------------------------
// Attention block for MI455X (gfx1250, wave32, WMMA, TDM).
// B=2, S=2048, HIST=2048, H=HK=16, D=128, DIM=2048, L=4096.
// Compute-bound (AI ~820 FLOP/B): bf16 WMMA, f32 accumulate, flash softmax.
// bf16 pre-convert of x/weights (GEMMs move raw bf16 bytes), TDM
// tensor_load_to_lds staging with LDS padding (6-arg builtin on this
// toolchain), double-buffered GEMM tiles (1 barrier/k-step), shift/mask remap.
// ---------------------------------------------------------------------------

#define BQ     2
#define SQ     2048
#define HISTQ  2048
#define HQ     16
#define DHEAD  128
#define DIMQ   2048
#define LTOT   (HISTQ + SQ)

typedef __attribute__((ext_vector_type(16))) __bf16 v16bf;
typedef __attribute__((ext_vector_type(8)))  float  v8f;

struct U8 { unsigned int x[8]; };

#if defined(__has_builtin)
#if __has_builtin(__builtin_amdgcn_tensor_load_to_lds) && __has_builtin(__builtin_amdgcn_s_wait_tensorcnt)
#define USE_TDM 1
#endif
#endif
#ifndef USE_TDM
#define USE_TDM 0
#endif

__device__ __forceinline__ unsigned short f2bf(float f) {
  unsigned int u = __builtin_bit_cast(unsigned int, f);
  u += 0x7FFFu + ((u >> 16) & 1u);          // round-to-nearest-even
  return (unsigned short)(u >> 16);
}

// A-fragment (16x32 bf16 MxK) from LDS row-major [row][k], ld even (ISA 7.12.2).
__device__ __forceinline__ v16bf ldsA(const unsigned short* p0, int row, int ld,
                                      int kBase, int lane) {
  const unsigned int* p = (const unsigned int*)(p0 + row * ld);
  int kb = (lane >> 4) * 16;
  U8 t;
#pragma unroll
  for (int v = 0; v < 8; ++v) {
    int k0 = kBase + kb + v * 2;
    t.x[v] = p[k0 >> 1];
  }
  return __builtin_bit_cast(v16bf, t);
}

// B-fragment (32x16 bf16 KxN); LDS holds B^T row-major [n][k]; lane-half = K16.
__device__ __forceinline__ v16bf ldsB(const unsigned short* p0, int n, int ld,
                                      int kBase, int lane) {
  const unsigned int* p = (const unsigned int*)(p0 + n * ld + kBase + ((lane >> 4) * 16));
  U8 t;
#pragma unroll
  for (int v = 0; v < 8; ++v) t.x[v] = p[v];
  return __builtin_bit_cast(v16bf, t);
}

__device__ __forceinline__ unsigned lds_off(const void* p) {
  return (unsigned)(uintptr_t)p;             // low 32 bits of generic = LDS byte addr
}

#if USE_TDM
typedef unsigned int v4u __attribute__((ext_vector_type(4)));
typedef int          v8i __attribute__((ext_vector_type(8)));
typedef int          v4i __attribute__((ext_vector_type(4)));

// 2D bf16 tile load via Tensor Data Mover. Tile tileRows x tileCols (elements),
// tensor row pitch = pitch elements. LDS gets pad of 4 DWORDs every 16 DWORDs
// (row 64B -> pitch 80B = 40 ushorts, matching the fragment loaders).
__device__ __forceinline__ void tdm_load_tile(const unsigned short* gsrc,
                                              unsigned ldsByteOff,
                                              int tensorRows, int pitch,
                                              int tileRows, int tileCols) {
  unsigned long long ga = (unsigned long long)(uintptr_t)gsrc;
  v4u g0;
  g0.x = 1u;                                         // count=1 (valid descriptor)
  g0.y = ldsByteOff;                                 // lds_addr
  g0.z = (unsigned)(ga & 0xFFFFFFFFu);               // global_addr[31:0]
  g0.w = (unsigned)((ga >> 32) & 0x01FFFFFFu) | (2u << 30);   // addr[56:32] | type=2
  v8i g1;
  g1[0] = (1 << 16)                                  // data_size: 2 bytes
        | (1 << 20)                                  // pad_enable
        | (3 << 22)                                  // pad_interval: 16 DWORDs
        | (3 << 25);                                 // pad_amount:   4 DWORDs
  g1[1] = (pitch & 0xFFFF) << 16;                    // tensor_dim0[15:0]
  g1[2] = ((pitch >> 16) & 0xFFFF) | ((tensorRows & 0xFFFF) << 16);
  g1[3] = ((tensorRows >> 16) & 0xFFFF) | (tileCols << 16);    // tile_dim0
  g1[4] = tileRows & 0xFFFF;                         // tile_dim1 (tile_dim2=0)
  g1[5] = pitch;                                     // tensor_dim0_stride[31:0]
  g1[6] = 0;
  g1[7] = 0;
  v4i g2 = {0, 0, 0, 0}, g3 = {0, 0, 0, 0};
  v8i g4 = {0, 0, 0, 0, 0, 0, 0, 0};                 // unused ext group (2D tile)
  // clang-23 / therock-10.0 arity: (v4u, v8i, v4i, v4i, v8i, i32 cpol)
  __builtin_amdgcn_tensor_load_to_lds(g0, g1, g2, g3, g4, 0);
}
#endif

// fallback manual staging: one uint4 (8 bf16) per thread, tile 64x32 bf16
__device__ __forceinline__ void stage_tile(const unsigned short* __restrict__ g,
                                           int pitch, unsigned short* dst, int tid) {
  int row = tid >> 2, seg = tid & 3;
  const uint4* src = (const uint4*)(g + (size_t)row * pitch + seg * 8);
  *((uint4*)(dst + row * 40 + seg * 8)) = *src;
}

// ---------------------------------------------------------------------------
// C = A(MxK,bf16) * W(NxK,bf16)^T. OUT_BF16: 0 -> f32 C, 1 -> bf16 C (x scale).
// Row remap gr = (m>>11)*obs + oroff + (m&2047) writes xk/xv into k_all/v_all.
// Tile 64x64, k-step 32, double-buffered LDS, 8 waves as 4(M)x2(N).
// ---------------------------------------------------------------------------
template<int OUT_BF16>
__global__ __launch_bounds__(256) void wmma_gemm_bf16(
    const unsigned short* __restrict__ A, const unsigned short* __restrict__ W,
    void* __restrict__ Cout, int M, int N, int K, int obs, int oroff, float oscale) {
  const int LDA = 40;                                // 64B row + 16B pad
  __shared__ __align__(16) unsigned short As[2][64 * 40];
  __shared__ __align__(16) unsigned short Ws[2][64 * 40];

  const int tid  = threadIdx.x;
  const int lane = tid & 31, wave = tid >> 5;
  const int mw = wave >> 1, nw = wave & 1;
  const int hi = lane >> 4;
  const int m0 = blockIdx.y * 64, n0 = blockIdx.x * 64;

  const unsigned short* Atile = A + (size_t)m0 * K;
  const unsigned short* Wtile = W + (size_t)n0 * K;

  v8f acc0 = {}, acc1 = {};
  const int ksteps = K / 32;

  // prologue: tile 0 -> buffer 0
#if USE_TDM
  if (wave == 0) {
    tdm_load_tile(Atile, lds_off(&As[0][0]), M, K, 64, 32);
    tdm_load_tile(Wtile, lds_off(&Ws[0][0]), N, K, 64, 32);
    __builtin_amdgcn_s_wait_tensorcnt(0);
  }
#else
  stage_tile(Atile, K, &As[0][0], tid);
  stage_tile(Wtile, K, &Ws[0][0], tid);
#endif
  __syncthreads();

  for (int kt = 0; kt < ksteps; ++kt) {
    const int cur = kt & 1, nxt = cur ^ 1;
    if (kt + 1 < ksteps) {                           // overlap next-tile DMA
#if USE_TDM
      if (wave == 0) {
        tdm_load_tile(Atile + (kt + 1) * 32, lds_off(&As[nxt][0]), M, K, 64, 32);
        tdm_load_tile(Wtile + (kt + 1) * 32, lds_off(&Ws[nxt][0]), N, K, 64, 32);
      }
#else
      stage_tile(Atile + (kt + 1) * 32, K, &As[nxt][0], tid);
      stage_tile(Wtile + (kt + 1) * 32, K, &Ws[nxt][0], tid);
      if (kt + 2 < ksteps) {
        __builtin_prefetch(Atile + (size_t)(tid >> 2) * K + (kt + 2) * 32, 0, 1);
        __builtin_prefetch(Wtile + (size_t)(tid >> 2) * K + (kt + 2) * 32, 0, 1);
      }
#endif
    }

    v16bf aF = ldsA(&As[cur][0], mw * 16 + (lane & 15), LDA, 0, lane);
    v16bf b0 = ldsB(&Ws[cur][0], nw * 32 + (lane & 15), LDA, 0, lane);
    v16bf b1 = ldsB(&Ws[cur][0], nw * 32 + 16 + (lane & 15), LDA, 0, lane);
    acc0 = __builtin_amdgcn_wmma_f32_16x16x32_bf16(false, aF, false, b0, (short)0, acc0, false, false);
    acc1 = __builtin_amdgcn_wmma_f32_16x16x32_bf16(false, aF, false, b1, (short)0, acc1, false, false);

#if USE_TDM
    if (wave == 0 && kt + 1 < ksteps) __builtin_amdgcn_s_wait_tensorcnt(0);
#endif
    __syncthreads();
  }

#pragma unroll
  for (int r = 0; r < 8; ++r) {
    int m  = m0 + mw * 16 + r + 8 * hi;              // C/D layout: M = r + 8*half
    int gr = ((m >> 11) * obs) + oroff + (m & 2047); // S_loc = 2048 (shift/mask)
    size_t rb = (size_t)gr * N;
    int c0 = n0 + nw * 32 + (lane & 15), c1 = c0 + 16;
    if (OUT_BF16) {
      unsigned short* C = (unsigned short*)Cout;
      C[rb + c0] = f2bf(acc0[r] * oscale);
      C[rb + c1] = f2bf(acc1[r] * oscale);
    } else {
      float* C = (float*)Cout;
      C[rb + c0] = acc0[r];
      C[rb + c1] = acc1[r];
    }
  }
}

// ---------------------------------------------------------------------------
// Flash attention: block per (b, h, 64-query tile), 64-key blocks streamed.
// Q is bf16, pre-scaled by 1/sqrt(D) in the Q-projection epilogue.
// Causal mask analytic (k_pos <= HIST + q_idx) == reference mask tensor.
// ---------------------------------------------------------------------------
__global__ __launch_bounds__(256) void attn_fwd(
    const unsigned short* __restrict__ Qws, const float* __restrict__ Kall,
    const float* __restrict__ Vall, unsigned short* __restrict__ Ows) {
  const int LDQ = 136, LDV = 72, LDP = 72, LSC = 65;
  __shared__ __align__(16) unsigned short Qs[64 * 136];   // Q tile  (bf16, scaled)
  __shared__ __align__(16) unsigned short Ks[64 * 136];   // K block (bf16)
  __shared__ __align__(16) unsigned short Vt[128 * 72];   // V block transposed [d][key]
  __shared__ __align__(16) unsigned short Ps[64 * 72];    // softmax probs (bf16)
  __shared__ float Sc[64 * 65];                           // raw scores (f32)
  __shared__ float mrow[64], lrow[64], arow[64];

  const int tid  = threadIdx.x;
  const int lane = tid & 31, wave = tid >> 5;
  const int mw = wave >> 1, nw = wave & 1;               // 4(M) x 2(N) wave grid
  const int hi = lane >> 4;
  const int b = blockIdx.z, h = blockIdx.y, qt = blockIdx.x;

  const int lr = tid >> 2;                               // loader row 0..63
  const int lc = (tid & 3) * 32;                         // loader col base (of 128)

  { // Q tile: raw bf16 copy (4x uint4 per thread)
    const uint4* q4 = (const uint4*)(Qws + ((size_t)(b * SQ + qt * 64 + lr)) * DIMQ
                                     + h * DHEAD + lc);
    uint4* dst = (uint4*)(Qs + lr * LDQ + lc);
#pragma unroll
    for (int j = 0; j < 4; ++j) dst[j] = q4[j];
  }
  if (tid < 64) { mrow[tid] = -3e38f; lrow[tid] = 0.f; }
  __syncthreads();

  v8f o0 = {}, o1 = {}, o2 = {}, o3 = {};                // rows mw*16.., cols nw*64..

  for (int kb = 0; kb < LTOT / 64; ++kb) {
    const int kpos0 = kb * 64;
    { // K block (f32 from d_out) -> Ks [key][d] bf16
      const float4* k4 = (const float4*)(Kall + ((size_t)(b * LTOT + kpos0 + lr)) * DIMQ
                                         + h * DHEAD + lc);
      unsigned short* dst = Ks + lr * LDQ + lc;
#pragma unroll
      for (int j = 0; j < 8; ++j) {
        float4 v = k4[j];
        dst[j * 4 + 0] = f2bf(v.x); dst[j * 4 + 1] = f2bf(v.y);
        dst[j * 4 + 2] = f2bf(v.z); dst[j * 4 + 3] = f2bf(v.w);
      }
    }
    { // V block -> Vt [d][key] bf16 (transposed: PV B-fragments contiguous)
      const float4* v4 = (const float4*)(Vall + ((size_t)(b * LTOT + kpos0 + lr)) * DIMQ
                                         + h * DHEAD + lc);
#pragma unroll
      for (int j = 0; j < 8; ++j) {
        float4 v = v4[j];
        Vt[(lc + j * 4 + 0) * LDV + lr] = f2bf(v.x);
        Vt[(lc + j * 4 + 1) * LDV + lr] = f2bf(v.y);
        Vt[(lc + j * 4 + 2) * LDV + lr] = f2bf(v.z);
        Vt[(lc + j * 4 + 3) * LDV + lr] = f2bf(v.w);
      }
    }
    __syncthreads();

    // S = Q * K^T  (64x64, D=128 -> 4 bf16 WMMA k-steps)
    v8f s0 = {}, s1 = {};
#pragma unroll
    for (int dk = 0; dk < 4; ++dk) {
      v16bf aF  = ldsA(Qs, mw * 16 + (lane & 15), LDQ, dk * 32, lane);
      v16bf bk0 = ldsB(Ks, nw * 32 +      (lane & 15), LDQ, dk * 32, lane);
      v16bf bk1 = ldsB(Ks, nw * 32 + 16 + (lane & 15), LDQ, dk * 32, lane);
      s0 = __builtin_amdgcn_wmma_f32_16x16x32_bf16(false, aF, false, bk0, (short)0, s0, false, false);
      s1 = __builtin_amdgcn_wmma_f32_16x16x32_bf16(false, aF, false, bk1, (short)0, s1, false, false);
    }
#pragma unroll
    for (int r = 0; r < 8; ++r) {
      int qr   = mw * 16 + r + 8 * hi;
      int qpos = HISTQ + qt * 64 + qr;
      int c0   = nw * 32 + (lane & 15);
      int c1   = c0 + 16;
      Sc[qr * LSC + c0] = s0[r] + ((kpos0 + c0 <= qpos) ? 0.f : -1e9f);
      Sc[qr * LSC + c1] = s1[r] + ((kpos0 + c1 <= qpos) ? 0.f : -1e9f);
    }
    __syncthreads();

    // online softmax: thread t owns query row t
    if (tid < 64) {
      const float* srow = Sc + tid * LSC;
      float mo = mrow[tid], mn = mo;
#pragma unroll 8
      for (int j = 0; j < 64; ++j) mn = fmaxf(mn, srow[j]);
      float al = __expf(mo - mn);
      float sum = 0.f;
      unsigned short* prow = Ps + tid * LDP;
#pragma unroll 8
      for (int j = 0; j < 64; ++j) {
        float p = __expf(srow[j] - mn);
        sum += p;
        prow[j] = f2bf(p);
      }
      lrow[tid] = lrow[tid] * al + sum;
      mrow[tid] = mn;
      arow[tid] = al;
    }
    __syncthreads();

    // O = O*alpha + P * V   (64x128, 2 bf16 WMMA k-steps over 64 keys)
#pragma unroll
    for (int r = 0; r < 8; ++r) {
      float al = arow[mw * 16 + r + 8 * hi];
      o0[r] *= al; o1[r] *= al; o2[r] *= al; o3[r] *= al;
    }
#pragma unroll
    for (int ks = 0; ks < 2; ++ks) {
      v16bf aP  = ldsA(Ps, mw * 16 + (lane & 15), LDP, ks * 32, lane);
      v16bf bv0 = ldsB(Vt, nw * 64 +      (lane & 15), LDV, ks * 32, lane);
      v16bf bv1 = ldsB(Vt, nw * 64 + 16 + (lane & 15), LDV, ks * 32, lane);
      v16bf bv2 = ldsB(Vt, nw * 64 + 32 + (lane & 15), LDV, ks * 32, lane);
      v16bf bv3 = ldsB(Vt, nw * 64 + 48 + (lane & 15), LDV, ks * 32, lane);
      o0 = __builtin_amdgcn_wmma_f32_16x16x32_bf16(false, aP, false, bv0, (short)0, o0, false, false);
      o1 = __builtin_amdgcn_wmma_f32_16x16x32_bf16(false, aP, false, bv1, (short)0, o1, false, false);
      o2 = __builtin_amdgcn_wmma_f32_16x16x32_bf16(false, aP, false, bv2, (short)0, o2, false, false);
      o3 = __builtin_amdgcn_wmma_f32_16x16x32_bf16(false, aP, false, bv3, (short)0, o3, false, false);
    }
    __syncthreads();
  }

#pragma unroll
  for (int r = 0; r < 8; ++r) {
    int qr = mw * 16 + r + 8 * hi;
    float inv = 1.0f / lrow[qr];
    unsigned short* ob = Ows + ((size_t)(b * SQ + qt * 64 + qr)) * DIMQ
                       + h * DHEAD + nw * 64 + (lane & 15);
    ob[0]  = f2bf(o0[r] * inv);
    ob[16] = f2bf(o1[r] * inv);
    ob[32] = f2bf(o2[r] * inv);
    ob[48] = f2bf(o3[r] * inv);
  }
}

// f32 -> bf16 elementwise (float4 -> 4x bf16 as uint2)
__global__ void cvt_bf16(const float4* __restrict__ src, unsigned short* __restrict__ dst,
                         long n4) {
  long i = (long)blockIdx.x * blockDim.x + threadIdx.x;
  if (i >= n4) return;
  float4 v = src[i];
  uint2 o;
  o.x = (unsigned)f2bf(v.x) | ((unsigned)f2bf(v.y) << 16);
  o.y = (unsigned)f2bf(v.z) | ((unsigned)f2bf(v.w) << 16);
  ((uint2*)dst)[i] = o;
}

// prev_k/prev_v -> history slots of k_all/v_all (float4 granularity)
__global__ void copy_hist4(const float4* __restrict__ src, float4* __restrict__ dst,
                           long total4, long per4, long stride4) {
  long i = (long)blockIdx.x * blockDim.x + threadIdx.x;
  if (i >= total4) return;
  long b = i / per4, off = i - b * per4;
  dst[b * stride4 + off] = src[i];
}

extern "C" void kernel_launch(void* const* d_in, const int* in_sizes, int n_in,
                              void* d_out, int out_size, void* d_ws, size_t ws_size,
                              hipStream_t stream) {
  (void)in_sizes; (void)n_in; (void)out_size; (void)ws_size;
  const float* x      = (const float*)d_in[0];
  const float* prev_k = (const float*)d_in[1];
  const float* prev_v = (const float*)d_in[2];
  /* d_in[3] = mask: reproduced analytically inside attn_fwd */
  const float* wq = (const float*)d_in[4];
  const float* wk = (const float*)d_in[5];
  const float* wv = (const float*)d_in[6];
  const float* wo = (const float*)d_in[7];

  float* out  = (float*)d_out;                               // [B,S,DIM]
  float* kall = out  + (size_t)BQ * SQ * DIMQ;               // [B,L,HK,D]
  float* vall = kall + (size_t)BQ * LTOT * DIMQ;             // [B,L,HK,D]

  const size_t nX = (size_t)BQ * SQ * DIMQ;                  // 8.4M elems
  const size_t nW = (size_t)DIMQ * DIMQ;                     // 4.2M elems
  unsigned short* xbf  = (unsigned short*)d_ws;              // bf16 workspace (~84MB)
  unsigned short* wqbf = xbf  + nX;
  unsigned short* wkbf = wqbf + nW;
  unsigned short* wvbf = wkbf + nW;
  unsigned short* wobf = wvbf + nW;
  unsigned short* qbf  = wobf + nW;                          // xq  (bf16, pre-scaled)
  unsigned short* abf  = qbf  + nX;                          // attn out (bf16)

  // 0) one-shot f32 -> bf16 converts (x + 4 weights)
  cvt_bf16<<<(int)(nX / 4 / 256), 256, 0, stream>>>((const float4*)x,  xbf,  nX / 4);
  cvt_bf16<<<(int)(nW / 4 / 256), 256, 0, stream>>>((const float4*)wq, wqbf, nW / 4);
  cvt_bf16<<<(int)(nW / 4 / 256), 256, 0, stream>>>((const float4*)wk, wkbf, nW / 4);
  cvt_bf16<<<(int)(nW / 4 / 256), 256, 0, stream>>>((const float4*)wv, wvbf, nW / 4);
  cvt_bf16<<<(int)(nW / 4 / 256), 256, 0, stream>>>((const float4*)wo, wobf, nW / 4);

  // 1) KV history copy
  long per4 = (long)HISTQ * DIMQ / 4;
  long total4 = (long)BQ * per4;
  long stride4 = (long)LTOT * DIMQ / 4;
  int cpB = (int)((total4 + 255) / 256);
  copy_hist4<<<cpB, 256, 0, stream>>>((const float4*)prev_k, (float4*)kall, total4, per4, stride4);
  copy_hist4<<<cpB, 256, 0, stream>>>((const float4*)prev_v, (float4*)vall, total4, per4, stride4);

  // 2) QKV projections (xq bf16 pre-scaled by 1/sqrt(D); xk/xv f32 -> k_all/v_all)
  dim3 g(DIMQ / 64, (BQ * SQ) / 64);
  wmma_gemm_bf16<1><<<g, 256, 0, stream>>>(xbf, wqbf, qbf,  BQ * SQ, DIMQ, DIMQ, SQ,   0,     0.08838834764831845f);
  wmma_gemm_bf16<0><<<g, 256, 0, stream>>>(xbf, wkbf, kall, BQ * SQ, DIMQ, DIMQ, LTOT, HISTQ, 1.0f);
  wmma_gemm_bf16<0><<<g, 256, 0, stream>>>(xbf, wvbf, vall, BQ * SQ, DIMQ, DIMQ, LTOT, HISTQ, 1.0f);

  // 3) flash attention (bf16 out)
  dim3 ga(SQ / 64, HQ, BQ);
  attn_fwd<<<ga, 256, 0, stream>>>(qbf, kall, vall, abf);

  // 4) output projection (f32 out)
  wmma_gemm_bf16<0><<<g, 256, 0, stream>>>(abf, wobf, out, BQ * SQ, DIMQ, DIMQ, SQ, 0, 1.0f);
}
